// FastMultiHeadedAttention_59906203844825
// MI455X (gfx1250) — matrix operand order repaired and checked by reading, not compiled, not run on hardware
//
#include <hip/hip_runtime.h>

// Problem constants (from reference): B=4, S=2048, D=512, H=8, R=8, dk=64
#define B_   4
#define S_   2048
#define D_   512
#define H_   8
#define R_   8
#define DK   64
#define NEDGE (B_*H_*2*R_*S_)   // 1,048,576 edges
#define QKV   (B_*H_*S_*DK)     // 4,194,304 floats per q/k/v/agg tensor (== B*S*D)
#define NDEN  (B_*H_*S_)        // 65,536 denominator slots

typedef __attribute__((ext_vector_type(16))) __bf16 v16bf;
typedef __attribute__((ext_vector_type(8)))  float  v8f;

union FragU { uint4 q[2]; v16bf v; };

// round-to-nearest-even f32 -> bf16 (bit manipulation; avoids __bf16 scalar ops)
__device__ __forceinline__ unsigned short f2bf(float f) {
  union { float f; unsigned u; } c; c.f = f;
  unsigned u = c.u;
  return (unsigned short)((u + 0x7FFFu + ((u >> 16) & 1u)) >> 16);
}

// async global->LDS 16B copy (per-lane), tracked by ASYNCcnt
__device__ __forceinline__ void async_copy16(unsigned lds_addr, const void* gaddr) {
  asm volatile("global_load_async_to_lds_b128 %0, %1, off"
               :: "v"(lds_addr), "v"((unsigned long long)(size_t)gaddr)
               : "memory");
}

// ---------------------------------------------------------------------------
// f32 -> bf16 bulk convert (8 elements / thread, 16B packed stores)
// ---------------------------------------------------------------------------
__global__ __launch_bounds__(256)
void f32_to_bf16_kernel(const float* __restrict__ src,
                        unsigned short* __restrict__ dst, int n8) {
  int i = blockIdx.x * 256 + threadIdx.x;
  if (i >= n8) return;
  const float4* s = (const float4*)src + (size_t)i * 2;
  float4 a = s[0], b = s[1];
  union { unsigned short h[8]; uint4 u; } pk;
  pk.h[0] = f2bf(a.x); pk.h[1] = f2bf(a.y); pk.h[2] = f2bf(a.z); pk.h[3] = f2bf(a.w);
  pk.h[4] = f2bf(b.x); pk.h[5] = f2bf(b.y); pk.h[6] = f2bf(b.z); pk.h[7] = f2bf(b.w);
  ((uint4*)dst)[i] = pk.u;
}

// agg [B,H,S,dk] f32  ->  [M=B*S, K=D] row-major bf16 (relayout + convert)
__global__ __launch_bounds__(256)
void agg_to_bf16_kernel(const float* __restrict__ agg,
                        unsigned short* __restrict__ dst) {
  int t = blockIdx.x * 256 + threadIdx.x;        // < 8192*512/8 = 524288
  int m  = t >> 6;                               // row (b,s)
  int kg = (t & 63) << 3;                        // k = kg..kg+7
  int b = m >> 11, s = m & (S_ - 1);
  int h = kg >> 6, d = kg & 63;
  const float* src = agg + ((((size_t)b * H_ + h) * S_ + s) << 6) + d;
  float4 a = *(const float4*)src;
  float4 c = *(const float4*)(src + 4);
  union { unsigned short hh[8]; uint4 u; } pk;
  pk.hh[0] = f2bf(a.x); pk.hh[1] = f2bf(a.y); pk.hh[2] = f2bf(a.z); pk.hh[3] = f2bf(a.w);
  pk.hh[4] = f2bf(c.x); pk.hh[5] = f2bf(c.y); pk.hh[6] = f2bf(c.z); pk.hh[7] = f2bf(c.w);
  *(uint4*)&dst[(size_t)m * D_ + kg] = pk.u;
}

// ---------------------------------------------------------------------------
// WMMA GEMM (bf16 inputs, f32 accumulate):  Y = A @ W^T + bias
//   A: [M,K] bf16 row-major, W: [N,K] bf16 row-major  (M=8192, N=K=512)
//   mode 0: Y written as [B,H,S,dk];  mode 1: Y is [M,N] row-major
// Tile 128x128x32, 8 waves, wave computes 64x32 (4x2 16x16 frags).
// Double-buffered LDS filled by global_load_async_to_lds_b128 (ASYNCcnt).
// ---------------------------------------------------------------------------
#define TM 128
#define TN 128
#define TK 32
#define KSTEPS (D_ / TK)   // 16

__global__ __launch_bounds__(256)
void wmma_gemm_async_kernel(const unsigned short* __restrict__ A,
                            const unsigned short* __restrict__ Wt,
                            const float* __restrict__ bias,
                            float* __restrict__ Y, int mode)
{
  __shared__ unsigned short lds[2][2][TM * TK];  // [buf][0=A,1=B][row*TK + k]

  const int n0    = blockIdx.x * TN;
  const int m0    = blockIdx.y * TM;
  const int tid   = threadIdx.x;
  const int lane  = tid & 31;
  const int wave  = tid >> 5;
  const int wr    = wave >> 2;        // 0..1
  const int wc    = wave & 3;         // 0..3
  const int lhalf = lane >> 4;        // 0/1
  const int l15   = lane & 15;

  v8f acc[4][2];
#pragma unroll
  for (int i = 0; i < 4; ++i)
#pragma unroll
    for (int j = 0; j < 2; ++j) {
      v8f z = {0.f, 0.f, 0.f, 0.f, 0.f, 0.f, 0.f, 0.f};
      acc[i][j] = z;
    }

  // issue one tile's async loads: 4 ops/thread (2 A + 2 B), 16B each
  auto issue_tile = [&](int kk, int p) {
#pragma unroll
    for (int i = 0; i < 2; ++i) {
      int slot = tid + i * 256;          // 0..511
      int row  = slot >> 2;              // 0..127
      int boff = (slot & 3) << 4;        // byte offset within 64B row
      unsigned la = (unsigned)(size_t)&lds[p][0][row * TK] + boff;
      async_copy16(la, (const char*)(A  + (size_t)(m0 + row) * D_ + kk) + boff);
      unsigned lb = (unsigned)(size_t)&lds[p][1][row * TK] + boff;
      async_copy16(lb, (const char*)(Wt + (size_t)(n0 + row) * D_ + kk) + boff);
    }
  };

  issue_tile(0, 0);

  for (int step = 0; step < KSTEPS; ++step) {
    int p = step & 1;
    if (step + 1 < KSTEPS) {
      issue_tile((step + 1) * TK, p ^ 1);            // prefetch next tile
      asm volatile("s_wait_asynccnt 0x4" ::: "memory");  // tile `step` resident
    } else {
      asm volatile("s_wait_asynccnt 0x0" ::: "memory");
    }
    __syncthreads();   // all threads' tile-`step` loads done -> LDS tile complete

    // per-lane fragments (CDNA5 16-bit A layout: K runs 0-7 / 16-23 per half)
    FragU fa[4], fb[2];
#pragma unroll
    for (int i = 0; i < 4; ++i) {
      int row = wr * 64 + i * 16 + l15;
      const unsigned short* pa = &lds[p][0][row * TK + lhalf * 8];
      fa[i].q[0] = *(const uint4*)pa;         // K = koff..koff+7
      fa[i].q[1] = *(const uint4*)(pa + 16);  // K = koff+16..koff+23
    }
#pragma unroll
    for (int j = 0; j < 2; ++j) {
      int col = wc * 32 + j * 16 + l15;
      const unsigned short* pb = &lds[p][1][col * TK + lhalf * 8];
      fb[j].q[0] = *(const uint4*)pb;         // K = koff..koff+7
      fb[j].q[1] = *(const uint4*)(pb + 16);  // K = koff+16..koff+23
    }

#pragma unroll
    for (int i = 0; i < 4; ++i)
#pragma unroll
      for (int j = 0; j < 2; ++j)
        acc[i][j] = __builtin_amdgcn_wmma_f32_16x16x32_bf16(
            false, fa[i].v, false, fb[j].v, (short)0, acc[i][j], false, false);

    __syncthreads();   // done reading buf p; safe target for issue at step+2
  }

  // epilogue: bias add + layout-aware store
  // C/D layout: VGPR r holds (M = base + lhalf*8 + r, N = base + l15)
#pragma unroll
  for (int i = 0; i < 4; ++i)
#pragma unroll
    for (int j = 0; j < 2; ++j) {
      int rowb = m0 + wr * 64 + i * 16 + lhalf * 8;
      int colb = n0 + wc * 32 + j * 16 + l15;
      float bv = bias[colb];
#pragma unroll
      for (int r = 0; r < 8; ++r) {
        int m = rowb + r;
        float val = acc[i][j][r] + bv;
        if (mode == 0) {
          int b = m >> 11, s = m & (S_ - 1);
          int h = colb >> 6, d = colb & 63;
          Y[((((size_t)b * H_ + h) * S_ + s) << 6) + d] = val;
        } else {
          Y[(size_t)m * D_ + colb] = val;
        }
      }
    }
}

// ---------------------------------------------------------------------------
// Edge scoring: 1 thread per edge; exp(masked score) + scatter-add denominator.
// q/k are [B,H,S,dk] f32 (L2-resident: 16.8 MB each).
// ---------------------------------------------------------------------------
__global__ __launch_bounds__(256)
void edge_score_kernel(const float* __restrict__ q, const float* __restrict__ k,
                       const int* __restrict__ snod, const int* __restrict__ enod,
                       const float* __restrict__ relq, const float* __restrict__ relk,
                       float* __restrict__ sflat, float* __restrict__ denom)
{
  int e  = blockIdx.x * 256 + threadIdx.x;       // < NEDGE
  int s  = e & (S_ - 1);
  int r2 = (e >> 11) & 15;
  int h  = (e >> 15) & 7;
  int b  = e >> 18;
  int rm = r2 & 7;                               // r2 % R
  int nidx = ((b * H_ + h) * R_ + rm) * S_ + s;
  int sv = snod[nidx], ev = enod[nidx];
  bool masked = (sv == -1) || (r2 == 0);
  int sz = (sv == -1) ? 0 : sv;
  int qi = (r2 < R_) ? ev : sz;                  // qidx = [end, start]
  int ki = (r2 < R_) ? sz : ev;                  // kidx = [start, end]
  int bh = b * H_ + h;

  const float* qp = q + ((size_t)(bh * S_ + qi) << 6);
  const float* kp = k + ((size_t)(bh * S_ + ki) << 6);
  const float* rq = relq + ((h * 16 + r2) << 6);
  const float* rk = relk + ((h * 16 + r2) << 6);

  float acc = 0.f;
#pragma unroll
  for (int d = 0; d < DK; d += 4) {
    float4 a = *(const float4*)(qp + d);
    float4 c = *(const float4*)(kp + d);
    float4 x = *(const float4*)(rq + d);
    float4 y = *(const float4*)(rk + d);
    // q.k + q.rel_k + rel_q.k == q.(k+rel_k) + rel_q.k
    acc += a.x * (c.x + y.x) + a.y * (c.y + y.y)
         + a.z * (c.z + y.z) + a.w * (c.w + y.w)
         + x.x * c.x + x.y * c.y + x.z * c.z + x.w * c.w;
  }
  // scale = 3 * sqrt(64) = 24;  exp(-1e9) underflows to 0 -> masked = 0
  float sf = masked ? 0.f : __expf(acc * (1.0f / 24.0f));
  sflat[e] = sf;
  atomicAdd(&denom[bh * S_ + qi], sf);
}

// ---------------------------------------------------------------------------
// Edge aggregation: 1 wave per edge, 2 dims per lane;
// agg[b,h,qi,:] += p * (v[b,h,ki,:] + rel_v[h,r2,:]) via f32 atomics.
// ---------------------------------------------------------------------------
__global__ __launch_bounds__(256)
void edge_aggregate_kernel(const float* __restrict__ v, const int* __restrict__ snod,
                           const int* __restrict__ enod, const float* __restrict__ relv,
                           const float* __restrict__ sflat, const float* __restrict__ denom,
                           float* __restrict__ agg)
{
  int t  = blockIdx.x * 256 + threadIdx.x;       // < NEDGE * 32
  int e  = t >> 5;
  int ln = t & 31;
  int s  = e & (S_ - 1);
  int r2 = (e >> 11) & 15;
  int h  = (e >> 15) & 7;
  int b  = e >> 18;
  int rm = r2 & 7;
  int nidx = ((b * H_ + h) * R_ + rm) * S_ + s;
  int sv = snod[nidx], ev = enod[nidx];
  int sz = (sv == -1) ? 0 : sv;
  int qi = (r2 < R_) ? ev : sz;
  int ki = (r2 < R_) ? sz : ev;
  int bh = b * H_ + h;

  float sf = sflat[e];
  float dn = denom[bh * S_ + qi];
  dn = (dn == 0.f) ? -1e9f : dn;
  float p = sf / dn;

  int d = ln << 1;
  float2 vv = *(const float2*)(v    + (((size_t)(bh * S_ + ki)) << 6) + d);
  float2 rv = *(const float2*)(relv + ((h * 16 + r2) << 6) + d);
  float* dst = agg + (((size_t)(bh * S_ + qi)) << 6) + d;
  atomicAdd(dst,     p * (vv.x + rv.x));
  atomicAdd(dst + 1, p * (vv.y + rv.y));
}

__global__ void zero_kernel(float* __restrict__ p, int n) {
  for (int i = blockIdx.x * blockDim.x + threadIdx.x; i < n;
       i += gridDim.x * blockDim.x)
    p[i] = 0.f;
}

// ---------------------------------------------------------------------------
extern "C" void kernel_launch(void* const* d_in, const int* in_sizes, int n_in,
                              void* d_out, int out_size, void* d_ws, size_t ws_size,
                              hipStream_t stream) {
  (void)in_sizes; (void)n_in; (void)out_size; (void)ws_size;

  const float* query = (const float*)d_in[0];
  const float* key_  = (const float*)d_in[1];
  const float* value = (const float*)d_in[2];
  const int*   snod  = (const int*)d_in[3];
  const int*   enod  = (const int*)d_in[4];
  const float* rel_q = (const float*)d_in[5];
  const float* rel_k = (const float*)d_in[6];
  const float* rel_v = (const float*)d_in[7];
  const float* Wq = (const float*)d_in[8];  const float* bq = (const float*)d_in[9];
  const float* Wk = (const float*)d_in[10]; const float* bk = (const float*)d_in[11];
  const float* Wv = (const float*)d_in[12]; const float* bv = (const float*)d_in[13];
  const float* Wo = (const float*)d_in[14]; const float* bo = (const float*)d_in[15];
  float* out = (float*)d_out;

  // workspace layout (f32): q | k | v | agg | sflat | denom | xb(bf16) | wb(bf16)
  float* ws    = (float*)d_ws;
  float* qd    = ws;
  float* kd    = ws + (size_t)QKV;
  float* vd    = ws + (size_t)2 * QKV;
  float* agg   = ws + (size_t)3 * QKV;
  float* sflat = ws + (size_t)4 * QKV;
  float* denom = sflat + NEDGE;
  unsigned short* xb = (unsigned short*)(denom + NDEN);   // 4,194,304 bf16 (8 MB)
  unsigned short* wb = xb + (size_t)QKV;                  // 262,144 bf16 (512 KB)

  dim3 gemmGrid(D_ / TN, (B_ * S_) / TM);   // (4, 64)
  const int XCONV_BLK = (QKV / 8) / 256;    // 2048
  const int WCONV_BLK = (D_ * D_ / 8) / 256;// 128

  // --- input projections -> [B,H,S,dk] (convert to bf16 once, reuse buffers) ---
  f32_to_bf16_kernel<<<XCONV_BLK, 256, 0, stream>>>(query, xb, QKV / 8);
  f32_to_bf16_kernel<<<WCONV_BLK, 256, 0, stream>>>(Wq, wb, D_ * D_ / 8);
  wmma_gemm_async_kernel<<<gemmGrid, 256, 0, stream>>>(xb, wb, bq, qd, 0);

  f32_to_bf16_kernel<<<XCONV_BLK, 256, 0, stream>>>(key_, xb, QKV / 8);
  f32_to_bf16_kernel<<<WCONV_BLK, 256, 0, stream>>>(Wk, wb, D_ * D_ / 8);
  wmma_gemm_async_kernel<<<gemmGrid, 256, 0, stream>>>(xb, wb, bk, kd, 0);

  f32_to_bf16_kernel<<<XCONV_BLK, 256, 0, stream>>>(value, xb, QKV / 8);
  f32_to_bf16_kernel<<<WCONV_BLK, 256, 0, stream>>>(Wv, wb, D_ * D_ / 8);
  wmma_gemm_async_kernel<<<gemmGrid, 256, 0, stream>>>(xb, wb, bv, vd, 0);

  // --- clear accumulation buffers (atomics accumulate across graph replays) ---
  zero_kernel<<<256, 256, 0, stream>>>(denom, NDEN);
  zero_kernel<<<1024, 256, 0, stream>>>(agg, QKV);

  // --- edge phase (L2-resident gathers + f32 atomic scatters) ---
  edge_score_kernel<<<NEDGE / 256, 256, 0, stream>>>(qd, kd, snod, enod,
                                                     rel_q, rel_k, sflat, denom);
  edge_aggregate_kernel<<<(NEDGE * 32) / 256, 256, 0, stream>>>(vd, snod, enod,
                                                                rel_v, sflat, denom, agg);

  // --- output projection: out = agg([B,S,D]) @ Wo^T + bo ---
  agg_to_bf16_kernel<<<XCONV_BLK, 256, 0, stream>>>(agg, xb);
  f32_to_bf16_kernel<<<WCONV_BLK, 256, 0, stream>>>(Wo, wb, D_ * D_ / 8);
  wmma_gemm_async_kernel<<<gemmGrid, 256, 0, stream>>>(xb, wb, bo, out, 1);
}
